// RVLinearlayer_841813590058
// MI455X (gfx1250) — matrix operand order repaired and checked by reading, not compiled, not run on hardware
//
#include <hip/hip_runtime.h>
#include <stdint.h>

typedef __attribute__((ext_vector_type(16))) __bf16 v16bf;
typedef __attribute__((ext_vector_type(8)))  float  v8f;
typedef __attribute__((ext_vector_type(4)))  int    v4i;

#define SIZE   4096
#define NELEM  (SIZE * SIZE)

#define BM     128
#define BN     64
#define KB     32
#define LDA    40                 // padded LDS stride (elements) for A tiles
#define LDBT   40                 // padded LDS stride (elements) for B tiles
#define KSTEPS (SIZE / KB)        // 128

#if __has_builtin(__builtin_amdgcn_global_load_async_to_lds_b128)
#define HAS_ASYNC_LDS 1
#else
#define HAS_ASYNC_LDS 0
#endif

// ---------- helpers ----------

__device__ __forceinline__ unsigned short f2bf(float f) {
  unsigned int x = __float_as_uint(f);
  x += 0x7fffu + ((x >> 16) & 1u);   // round-to-nearest-even
  return (unsigned short)(x >> 16);
}

__device__ __forceinline__ float softplusf(float x) {
  return (x > 20.0f) ? x : log1pf(expf(x));
}

// 16-byte-per-lane copy global -> LDS. Async (no data VGPRs, ASYNCcnt) when
// the CDNA5 builtin is available; transient synchronous copy otherwise.
__device__ __forceinline__ void cp16(const unsigned short* g, unsigned short* l) {
#if HAS_ASYNC_LDS
  __builtin_amdgcn_global_load_async_to_lds_b128(
      (__attribute__((address_space(1))) v4i*)(uintptr_t)g,
      (__attribute__((address_space(3))) v4i*)(uint32_t)(uintptr_t)l,
      0, 0);
#else
  *(uint4*)l = *(const uint4*)g;
#endif
}

__device__ __forceinline__ void cp16_wait() {
#if HAS_ASYNC_LDS
#if __has_builtin(__builtin_amdgcn_s_wait_asynccnt)
  __builtin_amdgcn_s_wait_asynccnt(0);
#else
  asm volatile("s_wait_asynccnt 0x0" ::: "memory");
#endif
#endif
}

// ---------- kernel 0: zero the two KL accumulators (ws is poisoned) ----------

__global__ void init_kernel(float* acc) {
  if (threadIdx.x < 2) acc[threadIdx.x] = 0.0f;
}

// ---------- kernel 1: mu -> bf16 {mu, mu^2}, row-major ----------

__global__ __launch_bounds__(256) void prep_mu_kernel(
    const float* __restrict__ mu_in,
    unsigned short* __restrict__ mu_bf, unsigned short* __restrict__ mu2_bf) {
  const int tid  = threadIdx.x;
  const int base = blockIdx.x * 2048;
#pragma unroll
  for (int j = 0; j < 8; ++j) {
    int i = base + j * 256 + tid;
    float mu = mu_in[i];
    mu_bf[i]  = f2bf(mu);
    mu2_bf[i] = f2bf(mu * mu);
  }
}

// ---------- kernel 2: w arrays -> bf16, TRANSPOSED ([N][K]) + KL sums ----------
// 64x64 tile transpose through LDS; coalesced loads and stores.

__global__ __launch_bounds__(256) void prep_w_kernel(
    const float* __restrict__ w_mu, const float* __restrict__ w_sigma,
    unsigned short* __restrict__ wmuT, unsigned short* __restrict__ spT,
    float* __restrict__ acc) {
  __shared__ unsigned short t1[64][65];
  __shared__ unsigned short t2[64][65];
  const int tid = threadIdx.x;
  const int bi  = blockIdx.x & 63;   // k-tile
  const int bj  = blockIdx.x >> 6;   // n-tile
  const int c   = tid & 63;
  const int r0  = tid >> 6;          // 0..3
  float l1 = 0.0f, l2 = 0.0f;
#pragma unroll 4
  for (int p = 0; p < 16; ++p) {
    int k  = r0 + p * 4;
    int gi = (bi * 64 + k) * SIZE + bj * 64 + c;   // [K][N] element
    float wm = w_mu[gi];
    t1[k][c] = f2bf(wm);
    l2 += fabsf(wm);
    float sp = softplusf(w_sigma[gi]);
    t2[k][c] = f2bf(sp);
    l1 += logf(sp) - sp;
  }
  __syncthreads();
#pragma unroll 4
  for (int p = 0; p < 16; ++p) {
    int n  = r0 + p * 4;
    int go = (bj * 64 + n) * SIZE + bi * 64 + c;   // [N][K] element
    wmuT[go] = t1[c][n];
    spT[go]  = t2[c][n];
  }
  __shared__ float r1[256];
  __shared__ float r2[256];
  r1[tid] = l1; r2[tid] = l2;
  __syncthreads();
  for (int s = 128; s > 0; s >>= 1) {
    if (tid < s) { r1[tid] += r1[tid + s]; r2[tid] += r2[tid + s]; }
    __syncthreads();
  }
  if (tid == 0) {
    atomicAdd(&acc[0], r1[0]);
    atomicAdd(&acc[1], r2[0]);
  }
}

// ---------- kernel 3: softplus(b_sigma) ----------

__global__ void bias_kernel(const float* __restrict__ b_sigma,
                            float* __restrict__ spb) {
  int i = blockIdx.x * 256 + threadIdx.x;
  if (i < SIZE) spb[i] = softplusf(b_sigma[i]);
}

// ---------- kernel 4: fused dual bf16 WMMA GEMM ----------
// C1 = A1 @ B1T^T + bias1 ; C2 = A2 @ B2T^T + bias2
// A row-major [M][K], B pre-transposed [N][K]; all staging is straight
// 16B-per-lane global->LDS copies (async on CDNA5).

union Frag { v16bf v; uint4 q[2]; };

__global__ __launch_bounds__(256) void gemm_fused_kernel(
    const unsigned short* __restrict__ A1, const unsigned short* __restrict__ A2,
    const unsigned short* __restrict__ B1T, const unsigned short* __restrict__ B2T,
    const float* __restrict__ bias1, const float* __restrict__ bias2,
    float* __restrict__ C1, float* __restrict__ C2) {
  __shared__ __align__(16) unsigned short As1[2][BM * LDA];
  __shared__ __align__(16) unsigned short As2[2][BM * LDA];
  __shared__ __align__(16) unsigned short Bt1[2][BN * LDBT];
  __shared__ __align__(16) unsigned short Bt2[2][BN * LDBT];

  const int tid  = threadIdx.x;
  const int lane = tid & 31;
  const int wave = tid >> 5;       // 0..7
  const int wm   = wave >> 1;      // 0..3 -> M offset (32 each)
  const int wn   = wave & 1;       // 0..1 -> N offset (32 each)
  const int blockM = blockIdx.x * BM;
  const int blockN = blockIdx.y * BN;

  // staging decomposition: 16B chunk per lane
  // A tile: 128 rows x 32 elems = 512 chunks -> tid and tid+256
  const int arow  = tid >> 2;                  // 0..63
  const int akoff = (tid & 3) * 8;             // 0,8,16,24
  const int aGlob = (blockM + arow) * SIZE + akoff;
  const int aLds  = arow * LDA + akoff;
  // B tile: 64 rows x 32 elems = 256 chunks -> tid
  const int bGlob = (blockN + arow) * SIZE + akoff;
  const int bLds  = arow * LDBT + akoff;

  auto stage = [&](int buf, int kk) {
    cp16(A1 + aGlob + kk,             &As1[buf][aLds]);
    cp16(A1 + aGlob + kk + 64 * SIZE, &As1[buf][aLds + 64 * LDA]);
    cp16(A2 + aGlob + kk,             &As2[buf][aLds]);
    cp16(A2 + aGlob + kk + 64 * SIZE, &As2[buf][aLds + 64 * LDA]);
    cp16(B1T + bGlob + kk, &Bt1[buf][bLds]);
    cp16(B2T + bGlob + kk, &Bt2[buf][bLds]);
  };

  v8f acc1[2][2] = {};
  v8f acc2[2][2] = {};

  const int fr_row = lane & 15;
  const int fr_hi  = lane >> 4;    // 0 or 1

  auto compute = [&](int buf) {
    Frag a1[2], a2[2], b1f[2], b2f[2];
#pragma unroll
    for (int mt = 0; mt < 2; ++mt) {
      // ISA 16-bit A 16x32 layout: lane row = lane%16,
      // chunks at K = hi*8..+7 and K = 16+hi*8..+7
      const unsigned short* p1 = &As1[buf][(wm * 32 + mt * 16 + fr_row) * LDA];
      a1[mt].q[0] = *(const uint4*)(p1 + fr_hi * 8);
      a1[mt].q[1] = *(const uint4*)(p1 + 16 + fr_hi * 8);
      const unsigned short* p2 = &As2[buf][(wm * 32 + mt * 16 + fr_row) * LDA];
      a2[mt].q[0] = *(const uint4*)(p2 + fr_hi * 8);
      a2[mt].q[1] = *(const uint4*)(p2 + 16 + fr_hi * 8);
    }
#pragma unroll
    for (int nt = 0; nt < 2; ++nt) {
      // ISA 16-bit B 32x16 layout: lane holds N = lane%16,
      // chunks at K = hi*8..+7 and K = 16+hi*8..+7 in the [N][K] LDS tile.
      const unsigned short* p1 =
          &Bt1[buf][(wn * 32 + nt * 16 + fr_row) * LDBT + fr_hi * 8];
      b1f[nt].q[0] = *(const uint4*)(p1);
      b1f[nt].q[1] = *(const uint4*)(p1 + 16);
      const unsigned short* p2 =
          &Bt2[buf][(wn * 32 + nt * 16 + fr_row) * LDBT + fr_hi * 8];
      b2f[nt].q[0] = *(const uint4*)(p2);
      b2f[nt].q[1] = *(const uint4*)(p2 + 16);
    }
#pragma unroll
    for (int mt = 0; mt < 2; ++mt)
#pragma unroll
      for (int nt = 0; nt < 2; ++nt) {
        acc1[mt][nt] = __builtin_amdgcn_wmma_f32_16x16x32_bf16(
            false, a1[mt].v, false, b1f[nt].v, (short)0, acc1[mt][nt], false, false);
        acc2[mt][nt] = __builtin_amdgcn_wmma_f32_16x16x32_bf16(
            false, a2[mt].v, false, b2f[nt].v, (short)0, acc2[mt][nt], false, false);
      }
  };

  stage(0, 0);
  cp16_wait();
  __syncthreads();
  int buf = 0;
  for (int ks = 0; ks < KSTEPS; ++ks) {
    if (ks + 1 < KSTEPS) stage(buf ^ 1, (ks + 1) * KB);   // overlaps with compute
    compute(buf);
    cp16_wait();
    __syncthreads();
    buf ^= 1;
  }

  // epilogue: D layout — VGPR r, lane L: row = r + 8*(L>>4), col = L&15
  const int col0    = blockN + wn * 32 + (lane & 15);
  const int rowbase = blockM + wm * 32 + 8 * (lane >> 4);
#pragma unroll
  for (int nt = 0; nt < 2; ++nt) {
    int col = col0 + nt * 16;
    float bv1 = bias1[col];
    float bv2 = bias2[col];
#pragma unroll
    for (int mt = 0; mt < 2; ++mt) {
      int rb = rowbase + mt * 16;
#pragma unroll
      for (int r = 0; r < 8; ++r) {
        C1[(rb + r) * SIZE + col] = acc1[mt][nt][r] + bv1;
        C2[(rb + r) * SIZE + col] = acc2[mt][nt][r] + bv2;
      }
    }
  }
}

// ---------- kernel 5: finalize KL scalar ----------
// kl = -0.5 * ( SIZE*mean(log sp - sp) - sum|w_mu| ) = -0.5 * (S1/4096 - S2)

__global__ void finalize_kernel(const float* __restrict__ acc,
                                float* __restrict__ kl_out) {
  if (threadIdx.x == 0) {
    float S1 = acc[0], S2 = acc[1];
    *kl_out = -0.5f * (S1 / 4096.0f - S2);
  }
}

// ---------- host launcher ----------

extern "C" void kernel_launch(void* const* d_in, const int* in_sizes, int n_in,
                              void* d_out, int out_size, void* d_ws, size_t ws_size,
                              hipStream_t stream) {
  (void)in_sizes; (void)n_in; (void)out_size; (void)ws_size;

  const float* mu_in   = (const float*)d_in[0];
  /* d_in[1] = sigma_in is unused by the reference */
  const float* w_mu    = (const float*)d_in[2];
  const float* w_sigma = (const float*)d_in[3];
  const float* b_mu    = (const float*)d_in[4];
  const float* b_sigma = (const float*)d_in[5];

  float* out       = (float*)d_out;
  float* mu_out    = out;
  float* sigma_out = out + (long long)NELEM;
  float* kl_out    = out + 2LL * NELEM;

  char* ws = (char*)d_ws;
  float* acc = (float*)ws;                       // 2 floats (KL partial sums)
  float* spb = (float*)(ws + 1024);              // 4096 floats: softplus(b_sigma)
  unsigned short* mu_bf  = (unsigned short*)(ws + (1 << 15));
  unsigned short* mu2_bf = mu_bf  + (long long)NELEM;
  unsigned short* wmuT   = mu2_bf + (long long)NELEM;   // [N][K] bf16
  unsigned short* spT    = wmuT   + (long long)NELEM;   // [N][K] bf16

  init_kernel<<<1, 64, 0, stream>>>(acc);
  prep_mu_kernel<<<NELEM / 2048, 256, 0, stream>>>(mu_in, mu_bf, mu2_bf);
  prep_w_kernel<<<4096, 256, 0, stream>>>(w_mu, w_sigma, wmuT, spT, acc);
  bias_kernel<<<16, 256, 0, stream>>>(b_sigma, spb);

  dim3 grid(SIZE / BM, SIZE / BN);
  gemm_fused_kernel<<<grid, 256, 0, stream>>>(mu_bf, mu2_bf, wmuT, spT,
                                              b_mu, spb, mu_out, sigma_out);

  finalize_kernel<<<1, 32, 0, stream>>>(acc, kl_out);
}
